// RegressionNet_14087492731445
// MI455X (gfx1250) — matrix operand order repaired and checked by reading, not compiled, not run on hardware
//
#include <hip/hip_runtime.h>

// ---------------------------------------------------------------------------
// CDNA5 (gfx1250) wave32 WMMA bf16 implementation of the regression net.
// All heavy math (cost volume, 3 convs, FC1) uses v_wmma_f32_16x16x32_bf16.
// ---------------------------------------------------------------------------

typedef __attribute__((ext_vector_type(16))) __bf16 v16bf;
typedef __attribute__((ext_vector_type(8)))  __bf16 v8bf;
typedef __attribute__((ext_vector_type(8)))  float  v8f;
typedef __attribute__((ext_vector_type(4)))  float  v4f;

union BFrag {
  v16bf v;
  v8bf  h[2];
  unsigned int u[8];
};

__device__ __forceinline__ unsigned short f2bf(float f) {
  unsigned u = __builtin_bit_cast(unsigned, f);
  u += 0x7FFFu + ((u >> 16) & 1u);       // round-to-nearest-even
  return (unsigned short)(u >> 16);
}

__device__ __forceinline__ v8f wmma_bf16(const BFrag& a, const BFrag& b, v8f c) {
  return __builtin_amdgcn_wmma_f32_16x16x32_bf16(
      /*neg_a=*/false, a.v, /*neg_b=*/false, b.v,
      /*c_mod=*/(short)0, c, /*reuse_a=*/false, /*reuse_b=*/false);
}

#define ZERO8F {0.f,0.f,0.f,0.f,0.f,0.f,0.f,0.f}

// Problem constants
#define BATCH 64
#define CCH   256        // feature channels
#define HW    256        // 16x16 pixels
#define DD    1089       // 33*33 cost channels
#define DDP   1120       // padded to multiple of 32
#define OUTC  512
#define MTOT  16384      // BATCH*HW rows
#define FCK   131072     // 512*256
#define FULLY 1024

// ---------------------------------------------------------------------------
// 1) L2 normalize over channels; write transposed [b][p][c] bf16 (both feats)
// ---------------------------------------------------------------------------
__global__ void k_normalize(const float* __restrict__ f1, const float* __restrict__ f2,
                            unsigned short* __restrict__ f1n, unsigned short* __restrict__ f2n) {
  int feat = blockIdx.x >> 6;
  int b    = blockIdx.x & 63;
  const float* src = (feat ? f2 : f1) + (size_t)b * (CCH * HW);
  unsigned short* dst = (feat ? f2n : f1n) + (size_t)b * (CCH * HW);
  int t = threadIdx.x;                 // t == pixel in phase 1, channel in phase 2
  __shared__ float s_inv[HW];
  float acc = 0.f;
  for (int c = 0; c < CCH; ++c) {
    float v = src[c * HW + t];
    acc += v * v;
  }
  s_inv[t] = 1.f / fmaxf(sqrtf(acc), 1e-12f);
  __syncthreads();
  for (int p = 0; p < HW; ++p)
    dst[p * CCH + t] = f2bf(src[t * HW + p] * s_inv[p]);
}

// ---------------------------------------------------------------------------
// 2) zero fill (used for padded cost tensor)
// ---------------------------------------------------------------------------
__global__ void k_fill_zero(unsigned int* __restrict__ p, int n) {
  int i = blockIdx.x * blockDim.x + threadIdx.x;
  if (i < n) p[i] = 0u;
}

// ---------------------------------------------------------------------------
// 3) Cost volume: per-batch 256x256x256 GEMM (WMMA) + leaky-ReLU scatter.
//    gc layout: [b][p1][DDP] bf16 (zero pre-filled; boundary shifts stay 0).
// ---------------------------------------------------------------------------
__global__ void __launch_bounds__(256) k_costvol(const unsigned short* __restrict__ f1n,
                                                 const unsigned short* __restrict__ f2n,
                                                 unsigned short* __restrict__ gc) {
  int wave = (blockIdx.x * blockDim.x + threadIdx.x) >> 5;
  int lane = threadIdx.x & 31;
  int b    = wave >> 8;                 // 256 tiles per batch
  int tile = wave & 255;
  int m0 = (tile >> 4) * 16;
  int n0 = (tile & 15) * 16;
  int la  = lane & 15;
  int kba = (lane < 16) ? 0 : 16;       // A-frag K base
  int kbb = (lane < 16) ? 0 : 16;       // B-frag K base
  const unsigned short* ap = f1n + (size_t)b * (HW * CCH) + (size_t)(m0 + la) * CCH + kba;
  const unsigned short* bp = f2n + (size_t)b * (HW * CCH) + (size_t)(n0 + la) * CCH + kbb;
  v8f acc = ZERO8F;
#pragma unroll
  for (int k0 = 0; k0 < CCH; k0 += 32) {
    BFrag a, bb;
    a.h[0]  = *(const v8bf*)(ap + k0);
    a.h[1]  = *(const v8bf*)(ap + k0 + 8);
    bb.h[0] = *(const v8bf*)(bp + k0);
    bb.h[1] = *(const v8bf*)(bp + k0 + 8);
    acc = wmma_bf16(a, bb, acc);
  }
  int p2 = n0 + la;
  int u = p2 >> 4, v = p2 & 15;
  int mrow = m0 + ((lane & 16) ? 8 : 0);
#pragma unroll
  for (int r = 0; r < 8; ++r) {
    int p1 = mrow + r;
    int i = p1 >> 4, j = p1 & 15;
    int dy = u - i + 16, dx = v - j + 16;     // always in [1,31] -> in-bounds
    float val = acc[r] * (1.f / 256.f);
    val = (val > 0.f) ? val : 0.01f * val;
    gc[((size_t)b * HW + p1) * DDP + dy * 33 + dx] = f2bf(val);
  }
}

// ---------------------------------------------------------------------------
// 4) Weight prep: OIHW f32 -> [o][tap][cin_padded] bf16 (zero padded)
// ---------------------------------------------------------------------------
__global__ void k_prep_w(const float* __restrict__ W, unsigned short* __restrict__ out,
                         int O, int I, int IP) {
  int n = O * 9 * IP;
  int idx = blockIdx.x * blockDim.x + threadIdx.x;
  int stride = gridDim.x * blockDim.x;
  for (; idx < n; idx += stride) {
    int c = idx % IP;
    int rem = idx / IP;
    int tap = rem % 9;
    int o = rem / 9;
    unsigned short v = 0;
    if (c < I) v = f2bf(W[((size_t)o * I + c) * 9 + tap]);
    out[idx] = v;
  }
}

// ---------------------------------------------------------------------------
// 5) Conv3x3 (pad 1, stride 1) as implicit GEMM with WMMA.
//    act: [b][p][CinP] bf16; wt: [Cout][9*CinP] bf16; y: [row][Cout] f32.
//    Block = 8 waves (4 M-groups x 2 N-groups); wave tile = 32x64
//    (2x4 WMMA tiles -> 8 WMMAs per 12 b128 loads per K-chunk).
// ---------------------------------------------------------------------------
__global__ void __launch_bounds__(256) k_conv(const unsigned short* __restrict__ act,
                                              const unsigned short* __restrict__ wt,
                                              float* __restrict__ y,
                                              int CinP, int Cout) {
  const int Kdim = 9 * CinP;
  int w = threadIdx.x >> 5;
  int lane = threadIdx.x & 31;
  int wm = w >> 1, wn = w & 1;
  int bm = blockIdx.x & 127;            // 128 M-blocks of 128 rows
  int bn = blockIdx.x >> 7;             // Cout/128 N-blocks of 128 cols
  int m_base = bm * 128 + wm * 32;
  int n_base = bn * 128 + wn * 64;
  int la  = lane & 15;
  int kba = (lane < 16) ? 0 : 16;
  int kbb = (lane < 16) ? 0 : 16;

  int bc[2], ic[2], jc[2];
#pragma unroll
  for (int mt = 0; mt < 2; ++mt) {
    int row = m_base + mt * 16 + la;
    int p = row & 255;
    bc[mt] = row >> 8;
    ic[mt] = p >> 4;
    jc[mt] = p & 15;
  }
  const unsigned short* wrow[4];
#pragma unroll
  for (int nt = 0; nt < 4; ++nt)
    wrow[nt] = wt + (size_t)(n_base + nt * 16 + la) * Kdim + kbb;

  v8f acc[2][4];
  {
    v8f z = ZERO8F;
#pragma unroll
    for (int mt = 0; mt < 2; ++mt)
#pragma unroll
      for (int nt = 0; nt < 4; ++nt) acc[mt][nt] = z;
  }

  int tap = 0, cin0 = 0;
  for (int k0 = 0; k0 < Kdim; k0 += 32) {
    int dyy = tap / 3 - 1, dxx = tap % 3 - 1;
    BFrag af[2];
#pragma unroll
    for (int mt = 0; mt < 2; ++mt) {
      int ii = ic[mt] + dyy, jj = jc[mt] + dxx;
      if ((unsigned)ii < 16u && (unsigned)jj < 16u) {
        const unsigned short* ap =
            act + ((size_t)(bc[mt] * HW + ii * 16 + jj) * CinP + cin0 + kba);
        af[mt].h[0] = *(const v8bf*)ap;
        af[mt].h[1] = *(const v8bf*)(ap + 8);
      } else {
#pragma unroll
        for (int q = 0; q < 8; ++q) af[mt].u[q] = 0u;
      }
    }
#pragma unroll
    for (int nt = 0; nt < 4; ++nt) {
      BFrag bf;
      const unsigned short* bp = wrow[nt] + k0;
      bf.h[0] = *(const v8bf*)bp;
      bf.h[1] = *(const v8bf*)(bp + 8);
#pragma unroll
      for (int mt = 0; mt < 2; ++mt)
        acc[mt][nt] = wmma_bf16(af[mt], bf, acc[mt][nt]);
    }
    cin0 += 32;
    if (cin0 == CinP) { cin0 = 0; ++tap; }
  }

  int mro = (lane & 16) ? 8 : 0;
#pragma unroll
  for (int mt = 0; mt < 2; ++mt)
#pragma unroll
    for (int nt = 0; nt < 4; ++nt) {
      int n = n_base + nt * 16 + la;
#pragma unroll
      for (int r = 0; r < 8; ++r) {
        int row = m_base + mt * 16 + mro + r;
        y[(size_t)row * Cout + n] = acc[mt][nt][r];
      }
    }
}

// ---------------------------------------------------------------------------
// 6) BatchNorm stats over all 16384 rows per channel (conv bias cancels in BN)
// ---------------------------------------------------------------------------
__global__ void k_stats(const float* __restrict__ y, float* __restrict__ mu,
                        float* __restrict__ rsig, int Cout) {
  int t = threadIdx.x;
  int clane = t & 63;
  int rg = t >> 6;
  int c = blockIdx.x * 64 + clane;
  float s1 = 0.f, s2 = 0.f;
  for (int r = rg; r < MTOT; r += 4) {
    float v = y[(size_t)r * Cout + c];
    s1 += v;
    s2 += v * v;
  }
  __shared__ float a1[256], a2[256];
  a1[t] = s1; a2[t] = s2;
  __syncthreads();
  if (t < 64) {
    float t1 = a1[t] + a1[t + 64] + a1[t + 128] + a1[t + 192];
    float t2 = a2[t] + a2[t + 64] + a2[t + 128] + a2[t + 192];
    float m = t1 * (1.f / (float)MTOT);
    float var = t2 * (1.f / (float)MTOT) - m * m;
    mu[c] = m;
    rsig[c] = rsqrtf(var + 1e-5f);
  }
}

// ---------------------------------------------------------------------------
// 7) BN apply + ReLU -> bf16 act [b][p][c]; optionally also transposed
//    actT [b][c*256+p] (the FC1 k-order) for layer 3.
// ---------------------------------------------------------------------------
__global__ void k_bn_apply(const float* __restrict__ y, const float* __restrict__ g,
                           const float* __restrict__ be, const float* __restrict__ mu,
                           const float* __restrict__ rsig,
                           unsigned short* __restrict__ act,
                           unsigned short* __restrict__ actT) {
  const int n = MTOT * OUTC;
  int idx = blockIdx.x * blockDim.x + threadIdx.x;
  int stride = gridDim.x * blockDim.x;
  for (; idx < n; idx += stride) {
    int c = idx & (OUTC - 1);
    int bp = idx >> 9;
    float v = (y[idx] - mu[c]) * rsig[c] * g[c] + be[c];
    v = fmaxf(v, 0.f);
    unsigned short h = f2bf(v);
    if (act) act[idx] = h;
    if (actT) {
      int b = bp >> 8, p = bp & 255;
      actT[(size_t)b * FCK + c * HW + p] = h;
    }
  }
}

// ---------------------------------------------------------------------------
// 8) FC1 as WMMA GEMM: M=64 (batch), N=1024, K=131072; Wf1 streamed f32 ->
//    bf16 in-register; K split into 32 segments -> deterministic partials.
// ---------------------------------------------------------------------------
__global__ void __launch_bounds__(256) k_fc1(const unsigned short* __restrict__ actT,
                                             const float* __restrict__ Wf1,
                                             float* __restrict__ part) {
  int wave = (blockIdx.x * blockDim.x + threadIdx.x) >> 5;
  int lane = threadIdx.x & 31;
  int kseg  = wave & 31;
  int ntile = wave >> 5;           // 0..63
  int la  = lane & 15;
  int kba = (lane < 16) ? 0 : 16;
  int kbb = (lane < 16) ? 0 : 16;
  int n = ntile * 16 + la;
  const float* wp0 = Wf1 + (size_t)n * FCK + (size_t)kseg * 4096 + kbb;
  const unsigned short* arow[4];
#pragma unroll
  for (int mt = 0; mt < 4; ++mt)
    arow[mt] = actT + (size_t)(mt * 16 + la) * FCK + (size_t)kseg * 4096 + kba;

  v8f acc[4];
  v8f z = ZERO8F;
  acc[0] = z; acc[1] = z; acc[2] = z; acc[3] = z;

  for (int k0 = 0; k0 < 4096; k0 += 32) {
    const float* wp = wp0 + k0;
    __builtin_prefetch(wp + 256, 0, 1);          // stream next weight lines
    union { v4f v; float f[4]; } q[4];
    q[0].v = *(const v4f*)(wp);
    q[1].v = *(const v4f*)(wp + 4);
    q[2].v = *(const v4f*)(wp + 8);
    q[3].v = *(const v4f*)(wp + 12);
    BFrag bf;
    unsigned short* hp = (unsigned short*)&bf;
#pragma unroll
    for (int i = 0; i < 16; ++i) hp[i] = f2bf(q[i >> 2].f[i & 3]);
#pragma unroll
    for (int mt = 0; mt < 4; ++mt) {
      BFrag af;
      const unsigned short* ap = arow[mt] + k0;
      af.h[0] = *(const v8bf*)ap;
      af.h[1] = *(const v8bf*)(ap + 8);
      acc[mt] = wmma_bf16(af, bf, acc[mt]);
    }
  }
  int mro = (lane & 16) ? 8 : 0;
#pragma unroll
  for (int mt = 0; mt < 4; ++mt)
#pragma unroll
    for (int r = 0; r < 8; ++r) {
      int b = mt * 16 + mro + r;
      part[((size_t)kseg * BATCH + b) * FULLY + n] = acc[mt][r];
    }
}

// ---------------------------------------------------------------------------
// 9) Reduce K-segments + bias + ReLU
// ---------------------------------------------------------------------------
__global__ void k_fc1red(const float* __restrict__ part, const float* __restrict__ bf1,
                         float* __restrict__ fc1y) {
  int idx = blockIdx.x * blockDim.x + threadIdx.x;   // 0..65535
  int b = idx >> 10, n = idx & (FULLY - 1);
  float s = bf1[n];
  for (int seg = 0; seg < 32; ++seg)
    s += part[((size_t)seg * BATCH + b) * FULLY + n];
  fc1y[idx] = fmaxf(s, 0.f);
}

// ---------------------------------------------------------------------------
// 10) FC2: (64 x 1024) @ (1024 x 8) + bias
// ---------------------------------------------------------------------------
__global__ void k_fc2(const float* __restrict__ fc1y, const float* __restrict__ Wf2,
                      const float* __restrict__ bf2, float* __restrict__ out) {
  int b = blockIdx.x;
  int t = threadIdx.x;
  __shared__ float red[256];
  float local[8];
#pragma unroll
  for (int o = 0; o < 8; ++o) local[o] = 0.f;
  for (int n = t; n < FULLY; n += 256) {
    float x = fc1y[b * FULLY + n];
#pragma unroll
    for (int o = 0; o < 8; ++o) local[o] += x * Wf2[o * FULLY + n];
  }
  for (int o = 0; o < 8; ++o) {
    red[t] = local[o];
    __syncthreads();
    for (int s = 128; s > 0; s >>= 1) {
      if (t < s) red[t] += red[t + s];
      __syncthreads();
    }
    if (t == 0) out[b * 8 + o] = red[0] + bf2[o];
    __syncthreads();
  }
}

// ---------------------------------------------------------------------------
// Launcher
// ---------------------------------------------------------------------------
extern "C" void kernel_launch(void* const* d_in, const int* in_sizes, int n_in,
                              void* d_out, int out_size, void* d_ws, size_t ws_size,
                              hipStream_t stream) {
  (void)in_sizes; (void)n_in; (void)out_size; (void)ws_size;
  const float* f1  = (const float*)d_in[0];
  const float* f2  = (const float*)d_in[1];
  const float* W1  = (const float*)d_in[2];
  const float* g1  = (const float*)d_in[4];
  const float* be1 = (const float*)d_in[5];
  const float* W2  = (const float*)d_in[6];
  const float* g2  = (const float*)d_in[8];
  const float* be2 = (const float*)d_in[9];
  const float* W3  = (const float*)d_in[10];
  const float* g3  = (const float*)d_in[12];
  const float* be3 = (const float*)d_in[13];
  const float* Wf1 = (const float*)d_in[14];
  const float* bf1 = (const float*)d_in[15];
  const float* Wf2 = (const float*)d_in[16];
  const float* bf2 = (const float*)d_in[17];
  float* out = (float*)d_out;

  // workspace carve (256B aligned). Total ~158 MB.
  char* ws = (char*)d_ws;
  auto carve = [&](size_t bytes) -> void* {
    void* p = (void*)ws;
    ws += (bytes + 255) & ~(size_t)255;
    return p;
  };
  unsigned short* f1n  = (unsigned short*)carve((size_t)BATCH * HW * CCH * 2);   // 8 MB
  unsigned short* f2n  = (unsigned short*)carve((size_t)BATCH * HW * CCH * 2);   // 8 MB
  unsigned short* gc   = (unsigned short*)carve((size_t)BATCH * HW * DDP * 2);   // 35 MB
  unsigned short* w1b  = (unsigned short*)carve((size_t)OUTC * 9 * DDP * 2);     // 9.8 MB
  unsigned short* w2b  = (unsigned short*)carve((size_t)OUTC * 9 * OUTC * 2);    // 4.5 MB
  unsigned short* w3b  = (unsigned short*)carve((size_t)OUTC * 9 * OUTC * 2);    // 4.5 MB
  float*          ybuf = (float*)carve((size_t)MTOT * OUTC * 4);                 // 32 MB
  unsigned short* act1 = (unsigned short*)carve((size_t)MTOT * OUTC * 2);        // 16 MB
  unsigned short* act2 = (unsigned short*)carve((size_t)MTOT * OUTC * 2);        // 16 MB
  unsigned short* actT = (unsigned short*)carve((size_t)BATCH * FCK * 2);        // 16 MB
  float*          mu   = (float*)carve(OUTC * 4);
  float*          rsig = (float*)carve(OUTC * 4);
  float*          part = (float*)carve((size_t)32 * BATCH * FULLY * 4);          // 8 MB
  float*          fc1y = (float*)carve((size_t)BATCH * FULLY * 4);

  // 1) normalize both features -> [b][p][c] bf16
  k_normalize<<<128, 256, 0, stream>>>(f1, f2, f1n, f2n);

  // 2) zero the padded cost tensor
  {
    int nwords = (int)((size_t)BATCH * HW * DDP * 2 / 4);
    k_fill_zero<<<(nwords + 255) / 256, 256, 0, stream>>>((unsigned int*)gc, nwords);
  }

  // 3) cost volume via per-batch WMMA GEMM
  k_costvol<<<2048, 256, 0, stream>>>(f1n, f2n, gc);

  // 4) weight prep
  k_prep_w<<<4096, 256, 0, stream>>>(W1, w1b, OUTC, DD, DDP);
  k_prep_w<<<4096, 256, 0, stream>>>(W2, w2b, OUTC, OUTC, OUTC);
  k_prep_w<<<4096, 256, 0, stream>>>(W3, w3b, OUTC, OUTC, OUTC);

  // 5) conv1 + BN + ReLU   (grid = 128 M-blocks * Cout/128 N-blocks)
  k_conv<<<128 * (OUTC / 128), 256, 0, stream>>>(gc, w1b, ybuf, DDP, OUTC);
  k_stats<<<OUTC / 64, 256, 0, stream>>>(ybuf, mu, rsig, OUTC);
  k_bn_apply<<<8192, 256, 0, stream>>>(ybuf, g1, be1, mu, rsig, act1, nullptr);

  // 6) conv2 + BN + ReLU
  k_conv<<<128 * (OUTC / 128), 256, 0, stream>>>(act1, w2b, ybuf, OUTC, OUTC);
  k_stats<<<OUTC / 64, 256, 0, stream>>>(ybuf, mu, rsig, OUTC);
  k_bn_apply<<<8192, 256, 0, stream>>>(ybuf, g2, be2, mu, rsig, act2, nullptr);

  // 7) conv3 + BN + ReLU (writes transposed layout for FC1 directly)
  k_conv<<<128 * (OUTC / 128), 256, 0, stream>>>(act2, w3b, ybuf, OUTC, OUTC);
  k_stats<<<OUTC / 64, 256, 0, stream>>>(ybuf, mu, rsig, OUTC);
  k_bn_apply<<<8192, 256, 0, stream>>>(ybuf, g3, be3, mu, rsig, nullptr, actT);

  // 8) FC1 (WMMA, K-split partials) -> reduce + ReLU -> FC2
  k_fc1<<<256, 256, 0, stream>>>(actT, Wf1, part);
  k_fc1red<<<256, 256, 0, stream>>>(part, bf1, fc1y);
  k_fc2<<<64, 256, 0, stream>>>(fc1y, Wf2, bf2, out);
}
